// SimMatrixEmbeddingLoss_84524956386054
// MI455X (gfx1250) — matrix operand order repaired and checked by reading, not compiled, not run on hardware
//
#include <hip/hip_runtime.h>

// ---------------------------------------------------------------------------
// SimMatrixEmbeddingLoss on MI455X (gfx1250):
//   loss = sum_{u,v} (1 - cos(n1_u, n2_v))^2 = U*V - 2*S1 + S2
//   S1 = sum(cos), S2 = sum(cos^2); cos = rownorm(in1) @ rownorm(in2)^T
// 4096x4096x1024 GEMM in bf16 WMMA (v_wmma_f32_16x16x32_bf16, f32 accum).
// B panel staged in LDS via async global->LDS copies (ASYNCcnt), double
// buffered so the DMA of chunk c+1 overlaps WMMA on chunk c.
// ---------------------------------------------------------------------------

typedef __bf16 bf16_t;
typedef bf16_t v16bf __attribute__((ext_vector_type(16)));
typedef bf16_t v8bf  __attribute__((ext_vector_type(8)));
typedef bf16_t v4bf  __attribute__((ext_vector_type(4)));
typedef float  v8f   __attribute__((ext_vector_type(8)));

#define P_DIM  1024
#define KC     128                 // K-chunk staged in LDS per buffer
#define NCHUNK (P_DIM / KC)        // 8
#define LDS_K  (KC + 8)            // pad: keep 16B align, spread LDS banks
#define EPSN   1e-8f

// ---------------- Phase 0: zero the accumulators (ws is poisoned) ----------
__global__ void sm_init(float* s) {
  s[0] = 0.0f;  // S1
  s[1] = 0.0f;  // S2
}

// ---------------- Phase 1: row L2-normalize, f32 -> bf16 -------------------
__global__ void sm_normalize_bf16(const float* __restrict__ in,
                                  bf16_t* __restrict__ out) {
  const int row  = blockIdx.x;
  const int t    = threadIdx.x;     // 0..255
  const int lane = t & 31;
  const int wv   = t >> 5;          // 8 waves (wave32)

  float4 v = ((const float4*)(in + (size_t)row * P_DIM))[t];
  float ss = v.x * v.x + v.y * v.y + v.z * v.z + v.w * v.w;
  #pragma unroll
  for (int off = 16; off > 0; off >>= 1)
    ss += __shfl_xor(ss, off, 32);

  __shared__ float red[8];
  if (lane == 0) red[wv] = ss;
  __syncthreads();
  float total = 0.0f;
  #pragma unroll
  for (int i = 0; i < 8; ++i) total += red[i];

  const float scale = 1.0f / fmaxf(sqrtf(total), EPSN);

  v4bf o;
  o.x = (bf16_t)(v.x * scale);
  o.y = (bf16_t)(v.y * scale);
  o.z = (bf16_t)(v.z * scale);
  o.w = (bf16_t)(v.w * scale);
  *(v4bf*)(out + (size_t)row * P_DIM + (size_t)t * 4) = o;
}

// Async 16B global -> LDS copy, tracked by ASYNCcnt (CDNA5 §10.7 / §15.18).
__device__ __forceinline__ void async_copy16(const bf16_t* gp, uint32_t lds_off) {
  asm volatile("global_load_async_to_lds_b128 %0, %1, off"
               :: "v"(lds_off), "v"(gp)
               : "memory");
}

// ---------------- Phase 2: WMMA GEMM + fused loss reduction ----------------
// Block = 256 threads = 8 waves, owns a 128x128 tile of the cosine matrix.
// Wave w (rg = w>>1, cg = w&1) computes a 32x64 sub-tile: 2 row-tiles x
// 4 col-tiles of 16x16 -> 8 v8f accumulators.
//
// bf16 fragment layouts per CDNA5 ISA 7.12.2 (wave32):
//   A 16x32: lane L -> row (L&15); elems 0-7 = K[(L>>4)*8..], 8-15 = K[16+(L>>4)*8..]
//   B 32x16: lane L -> col (L&15); elems 0-7 = K[(L>>4)*8..], 8-15 = K[16+(L>>4)*8..]
__global__ void __launch_bounds__(256)
sm_cos_loss_wmma(const bf16_t* __restrict__ n1,
                 const bf16_t* __restrict__ n2,
                 float* __restrict__ sums) {
  const int lane = threadIdx.x & 31;
  const int wv   = threadIdx.x >> 5;
  const int m    = lane & 15;
  const int kh   = lane >> 4;       // K-half covered by this lane
  const int rg   = wv >> 1;         // row-group 0..3
  const int cg   = wv & 1;          // col-group 0..1

  const int uBase = blockIdx.y * 128;
  const int vBase = blockIdx.x * 128;

  // Double-buffered B panel: rows = n2 rows [vBase, vBase+128), cols = K chunk
  __shared__ bf16_t sB[2][128][LDS_K];

  v8f acc0[4], acc1[4];
  #pragma unroll
  for (int j = 0; j < 4; ++j) { acc0[j] = {}; acc1[j] = {}; }

  const bf16_t* aRow0 =
      n1 + (size_t)(uBase + rg * 32 + m) * P_DIM;        // row-tile 0
  const bf16_t* aRow1 = aRow0 + (size_t)16 * P_DIM;      // row-tile 1

  const int t = threadIdx.x;

  // Issue async copy of one 128x128 bf16 chunk (32KB = 2048 x 16B segments;
  // 8 segments per thread, 8 async ops per wave -> ASYNCcnt granularity 8).
  auto issue_chunk = [&](int c, int buf) {
    const int k0 = c * KC;
    #pragma unroll
    for (int i = 0; i < 8; ++i) {
      const int seg  = i * 256 + t;       // 0..2047
      const int row  = seg >> 4;          // 16 segments per 128-elem row
      const int kseg = seg & 15;
      const bf16_t* gp =
          n2 + (size_t)(vBase + row) * P_DIM + k0 + kseg * 8;
      const uint32_t lp = (uint32_t)(uintptr_t)&sB[buf][row][kseg * 8];
      async_copy16(gp, lp);
    }
  };

  issue_chunk(0, 0);

  for (int c = 0; c < NCHUNK; ++c) {
    const int buf = c & 1;
    if (c + 1 < NCHUNK) {
      issue_chunk(c + 1, buf ^ 1);
      // 16 ops in flight; wait until only chunk c+1's 8 remain (in-order)
      asm volatile("s_wait_asynccnt 0x8" ::: "memory");
    } else {
      asm volatile("s_wait_asynccnt 0x0" ::: "memory");
    }
    __syncthreads();   // LDS buffer `buf` fully written for all waves

    const int k0 = c * KC;
    #pragma unroll
    for (int kk = 0; kk < KC; kk += 32) {
      const int k = k0 + kk;

      // A fragments from global (L1-resident, per-row reuse only 2x)
      v8bf a0lo = *(const v8bf*)(aRow0 + k + kh * 8);
      v8bf a0hi = *(const v8bf*)(aRow0 + k + 16 + kh * 8);
      v16bf aA  = __builtin_shufflevector(a0lo, a0hi,
                    0,1,2,3,4,5,6,7,8,9,10,11,12,13,14,15);
      v8bf a1lo = *(const v8bf*)(aRow1 + k + kh * 8);
      v8bf a1hi = *(const v8bf*)(aRow1 + k + 16 + kh * 8);
      v16bf aB  = __builtin_shufflevector(a1lo, a1hi,
                    0,1,2,3,4,5,6,7,8,9,10,11,12,13,14,15);

      #pragma unroll
      for (int j = 0; j < 4; ++j) {
        const int row = cg * 64 + j * 16 + m;   // B column = n2 row
        v8bf blo = *(const v8bf*)&sB[buf][row][kk + kh * 8];
        v8bf bhi = *(const v8bf*)&sB[buf][row][kk + 16 + kh * 8];
        v16bf b  = __builtin_shufflevector(blo, bhi,
                     0,1,2,3,4,5,6,7,8,9,10,11,12,13,14,15);
        acc0[j] = __builtin_amdgcn_wmma_f32_16x16x32_bf16(
            false, aA, false, b, (short)0, acc0[j], false, false);
        acc1[j] = __builtin_amdgcn_wmma_f32_16x16x32_bf16(
            false, aB, false, b, (short)0, acc1[j], false, false);
      }
    }
    __syncthreads();   // all waves done reading `buf` before next overwrite
  }

  // Fused reduction: S1 += c, S2 += c*c over this wave's 32x64 sub-tile
  float s1 = 0.0f, s2 = 0.0f;
  #pragma unroll
  for (int j = 0; j < 4; ++j) {
    #pragma unroll
    for (int e = 0; e < 8; ++e) {
      float c0 = acc0[j][e], c1 = acc1[j][e];
      s1 += c0 + c1;
      s2 += c0 * c0 + c1 * c1;
    }
  }
  #pragma unroll
  for (int off = 16; off > 0; off >>= 1) {
    s1 += __shfl_xor(s1, off, 32);
    s2 += __shfl_xor(s2, off, 32);
  }

  __shared__ float r1[8], r2[8];
  if (lane == 0) { r1[wv] = s1; r2[wv] = s2; }
  __syncthreads();
  if (t == 0) {
    float t1 = 0.0f, t2 = 0.0f;
    #pragma unroll
    for (int i = 0; i < 8; ++i) { t1 += r1[i]; t2 += r2[i]; }
    atomicAdd(&sums[0], t1);   // global_atomic_add_f32
    atomicAdd(&sums[1], t2);
  }
}

// ---------------- Phase 3: finalize loss in double -------------------------
__global__ void sm_finalize(const float* __restrict__ sums,
                            float* __restrict__ out, int U, int V) {
  double loss = (double)U * (double)V - 2.0 * (double)sums[0] + (double)sums[1];
  out[0] = (float)loss;
}

extern "C" void kernel_launch(void* const* d_in, const int* in_sizes, int n_in,
                              void* d_out, int out_size, void* d_ws, size_t ws_size,
                              hipStream_t stream) {
  (void)n_in; (void)out_size; (void)ws_size;
  const float* in1 = (const float*)d_in[0];
  const float* in2 = (const float*)d_in[1];
  const int U = in_sizes[0] / P_DIM;   // 4096
  const int V = in_sizes[1] / P_DIM;   // 4096

  // Workspace: [2 floats S1/S2][pad to 256B][n1 bf16 8MB][n2 bf16 8MB]
  char*   ws   = (char*)d_ws;
  float*  sums = (float*)ws;
  bf16_t* n1   = (bf16_t*)(ws + 256);
  bf16_t* n2   = (bf16_t*)(ws + 256 + (size_t)U * P_DIM * sizeof(bf16_t));

  sm_init<<<1, 1, 0, stream>>>(sums);
  sm_normalize_bf16<<<U, 256, 0, stream>>>(in1, n1);
  sm_normalize_bf16<<<V, 256, 0, stream>>>(in2, n2);

  dim3 grid(V / 128, U / 128);
  sm_cos_loss_wmma<<<grid, 256, 0, stream>>>(n1, n2, sums);

  sm_finalize<<<1, 1, 0, stream>>>(sums, (float*)d_out, U, V);
}
